// MultiHeadAttention_37701222924504
// MI455X (gfx1250) — matrix operand order repaired and checked by reading, not compiled, not run on hardware
//
#include <hip/hip_runtime.h>

// MHA: B=2, S=4096, D=512, H=8, DK=64.
// ~86 GFLOP vs ~150MB traffic -> compute bound -> bf16 WMMA (f32 accum).
// Register-blocked: GEMM waves own 64x64 tiles (16 WMMA per 16 B128 loads),
// attention waves own 32 query rows (16 WMMA per 16 B128 loads per key step).

#define B_  2
#define S_  4096
#define D_  512
#define H_  8
#define DK_ 64
#define BS_ (B_ * S_)   // 8192 rows

typedef __attribute__((ext_vector_type(16))) __bf16 v16bf;
typedef __attribute__((ext_vector_type(8)))  __bf16 v8bf;
typedef __attribute__((ext_vector_type(4)))  __bf16 v4bf;
typedef __attribute__((ext_vector_type(8)))  float  v8f;

static __device__ __forceinline__ v8f wmma_bf16(v16bf a, v16bf b, v8f c) {
  // D = A(16x32) * B(32x16) + C, f32 accumulate
  return __builtin_amdgcn_wmma_f32_16x16x32_bf16(false, a, false, b, (short)0, c,
                                                 false, false);
}

// A fragment: 16(M) x 32(K) bf16, from row-major memory with leading dim ld.
// ISA layout: lanes 0-15 hold K=[0..7,16..23], lanes 16-31 hold K=[8..15,24..31].
static __device__ __forceinline__ v16bf load_a(const __bf16* p, int ld) {
  const int lane = threadIdx.x & 31;
  const int row  = lane & 15;
  const int kb   = (lane & 16) ? 8 : 0;
  const __bf16* r = p + (size_t)row * ld + kb;
  union { v16bf f; v8bf h[2]; } u;
  u.h[0] = *(const v8bf*)(r);        // K = kb .. kb+7      -> VGPR 0..3
  u.h[1] = *(const v8bf*)(r + 16);   // K = kb+16 .. kb+23  -> VGPR 4..7
  return u.f;
}

// B fragment: 32(K) x 16(N). Memory holds B^T row-major ([N x K], ld).
// ISA layout: lanes 0-15 hold K=[0..7,16..23], lanes 16-31 hold K=[8..15,24..31], per column n.
static __device__ __forceinline__ v16bf load_b(const __bf16* p, int ld) {
  const int lane = threadIdx.x & 31;
  const int n    = lane & 15;
  const int kb   = (lane & 16) ? 8 : 0;
  const __bf16* r = p + (size_t)n * ld + kb;
  union { v16bf f; v8bf h[2]; } u;
  u.h[0] = *(const v8bf*)(r);        // K = kb .. kb+7
  u.h[1] = *(const v8bf*)(r + 16);   // K = kb+16 .. kb+23
  return u.f;
}

// ---------------- Stage 0: f32 -> bf16 conversion (vectorized x4) ----------
__global__ __launch_bounds__(256) void cvt_bf16(const float* __restrict__ in,
                                                __bf16* __restrict__ out, int n4) {
  const int i = blockIdx.x * blockDim.x + threadIdx.x;
  if (i < n4) {
    const float4 f = ((const float4*)in)[i];
    v4bf o;
    o.x = (__bf16)f.x; o.y = (__bf16)f.y; o.z = (__bf16)f.z; o.w = (__bf16)f.w;
    ((v4bf*)out)[i] = o;
  }
}

// ---------------- Stage 1/3: GEMM out = X @ W^T + bias ---------------------
// X: [BS_ x 512] bf16, W: [512 x 512] bf16 row-major (rows are output cols).
// One wave computes a 64x64 output tile: 4 A-frags x 4 B-frags -> 16 WMMAs
// per k-step of 32 (1 B128 vmem load per WMMA).
// mode 0: store bf16 [B,H,S,DK]   (Q, K heads)
// mode 1: store bf16 [B,H,DK,S]   (V transposed for P@V B-fragments)
// mode 2: store f32  [BS_ x 512]  (final output)
__global__ __launch_bounds__(256) void gemm512(const __bf16* __restrict__ X,
                                               const __bf16* __restrict__ W,
                                               const float*  __restrict__ bias,
                                               void* __restrict__ outp, int mode) {
  const int wid = blockIdx.x * (blockDim.x >> 5) + (threadIdx.x >> 5);
  const int NB  = D_ / 64;                       // 8 column blocks
  const int mb  = wid / NB;
  const int nb  = wid - mb * NB;
  if (mb >= BS_ / 64) return;
  const int lane = threadIdx.x & 31;

  const __bf16* Xa = X + (size_t)mb * 64 * D_;
  const __bf16* Wb = W + (size_t)nb * 64 * D_;

  v8f acc[4][4] = {};
  for (int k = 0; k < D_; k += 32) {
    v16bf a[4], b[4];
#pragma unroll
    for (int i = 0; i < 4; ++i) a[i] = load_a(Xa + (size_t)i * 16 * D_ + k, D_);
#pragma unroll
    for (int i = 0; i < 4; ++i) b[i] = load_b(Wb + (size_t)i * 16 * D_ + k, D_);
#pragma unroll
    for (int i = 0; i < 4; ++i)
#pragma unroll
      for (int jn = 0; jn < 4; ++jn)
        acc[i][jn] = wmma_bf16(a[i], b[jn], acc[i][jn]);
  }

  const int cn = lane & 15;
  const int mhalf = (lane & 16) ? 8 : 0;
#pragma unroll
  for (int jn = 0; jn < 4; ++jn) {
    const int n = nb * 64 + jn * 16 + cn;
    const float bv = bias[n];
    const int h = n >> 6, dk = n & 63;
#pragma unroll
    for (int i = 0; i < 4; ++i) {
      const int mbase = mb * 64 + i * 16 + mhalf;
#pragma unroll
      for (int v = 0; v < 8; ++v) {
        const float x = acc[i][jn][v] + bv;
        const int m = mbase + v;
        if (mode == 2) {
          ((float*)outp)[(size_t)m * D_ + n] = x;
        } else {
          const int b = m >> 12, s = m & (S_ - 1);
          const size_t idx = (mode == 0)
              ? ((((size_t)b * H_ + h) * S_  + s)  * DK_ + dk)
              : ((((size_t)b * H_ + h) * DK_ + dk) * S_  + s);
          ((__bf16*)outp)[idx] = (__bf16)x;
        }
      }
    }
  }
}

// ---------------- Stage 2: flash attention ---------------------------------
// One wave: 32 query rows (2 tiles) of one (b,h). Stream keys in tiles of 32.
// K/V fragments are reused across both query tiles: 16 WMMA per 16 B128 loads.
__global__ __launch_bounds__(256) void attn_fa(const __bf16* __restrict__ Qh,
                                               const __bf16* __restrict__ Kh,
                                               const __bf16* __restrict__ Vt,
                                               __bf16* __restrict__ Ctx) {
  __shared__ __bf16 Plds[8][2][16 * 32];         // wave-private P tiles (16 KB)
  const int w    = threadIdx.x >> 5;
  const int lane = threadIdx.x & 31;
  const int gw   = blockIdx.x * 8 + w;
  const int qb   = gw & (S_ / 32 - 1);           // query block 0..127
  const int bh   = gw >> 7;                      // (b*H + h) 0..15

  const __bf16* Qp = Qh + ((size_t)bh * S_ + (size_t)qb * 32) * DK_;
  const __bf16* Kp = Kh + (size_t)bh * S_ * DK_;
  const __bf16* Vp = Vt + (size_t)bh * DK_ * S_;

  v16bf q[2][2];
#pragma unroll
  for (int t = 0; t < 2; ++t) {
    q[t][0] = load_a(Qp + (size_t)t * 16 * DK_,      DK_);   // dk 0..31
    q[t][1] = load_a(Qp + (size_t)t * 16 * DK_ + 32, DK_);   // dk 32..63
  }

  v8f acc[2][4] = {};
  float mrow[2][8], lrow[2][8];
#pragma unroll
  for (int t = 0; t < 2; ++t)
#pragma unroll
    for (int v = 0; v < 8; ++v) { mrow[t][v] = -1e30f; lrow[t][v] = 0.f; }
  const float scale = 0.125f;                    // 1/sqrt(DK)

  const int cn = lane & 15;
  const int mh = (lane & 16) ? 8 : 0;

  for (int j = 0; j < S_; j += 32) {
    // K fragments for keys j..j+31 (shared by both query tiles)
    v16bf kf[4];
    kf[0] = load_b(Kp + (size_t)j * DK_,             DK_);
    kf[1] = load_b(Kp + (size_t)j * DK_ + 32,        DK_);
    kf[2] = load_b(Kp + (size_t)(j + 16) * DK_,      DK_);
    kf[3] = load_b(Kp + (size_t)(j + 16) * DK_ + 32, DK_);

    v8f s[2][2] = {};
#pragma unroll
    for (int t = 0; t < 2; ++t) {
      s[t][0] = wmma_bf16(q[t][0], kf[0], s[t][0]);
      s[t][0] = wmma_bf16(q[t][1], kf[1], s[t][0]);
      s[t][1] = wmma_bf16(q[t][0], kf[2], s[t][1]);
      s[t][1] = wmma_bf16(q[t][1], kf[3], s[t][1]);
    }

    // V fragments (shared by both query tiles)
    v16bf vf[4];
#pragma unroll
    for (int d = 0; d < 4; ++d)
      vf[d] = load_b(Vp + (size_t)d * 16 * S_ + j, S_);

#pragma unroll
    for (int t = 0; t < 2; ++t) {
      // online softmax: row max over 32 keys (butterfly over 16-lane halves)
      float tm[8];
#pragma unroll
      for (int v = 0; v < 8; ++v) {
        s[t][0][v] *= scale; s[t][1][v] *= scale;
        tm[v] = fmaxf(s[t][0][v], s[t][1][v]);
      }
#pragma unroll
      for (int off = 1; off < 16; off <<= 1) {
#pragma unroll
        for (int v = 0; v < 8; ++v)
          tm[v] = fmaxf(tm[v], __shfl_xor(tm[v], off, 32));
      }

      float rs[8];
#pragma unroll
      for (int v = 0; v < 8; ++v) {
        const float mn = fmaxf(mrow[t][v], tm[v]);
        const float al = __expf(mrow[t][v] - mn);
        mrow[t][v] = mn;
        s[t][0][v] = __expf(s[t][0][v] - mn);
        s[t][1][v] = __expf(s[t][1][v] - mn);
        rs[v] = s[t][0][v] + s[t][1][v];
        lrow[t][v] *= al;
        acc[t][0][v] *= al; acc[t][1][v] *= al;
        acc[t][2][v] *= al; acc[t][3][v] *= al;
      }
#pragma unroll
      for (int off = 1; off < 16; off <<= 1) {
#pragma unroll
        for (int v = 0; v < 8; ++v)
          rs[v] += __shfl_xor(rs[v], off, 32);
      }
#pragma unroll
      for (int v = 0; v < 8; ++v) lrow[t][v] += rs[v];

      // re-layout P: C-fragment -> LDS -> A-fragment (wave-private, DS in-order)
      __bf16* pl = &Plds[w][t][0];
#pragma unroll
      for (int v = 0; v < 8; ++v) {
        pl[(mh + v) * 32 + cn]      = (__bf16)s[t][0][v];
        pl[(mh + v) * 32 + 16 + cn] = (__bf16)s[t][1][v];
      }
      __asm__ volatile("" ::: "memory");
      const v16bf pa = load_a(pl, 32);

      // ctx += P(16x32) @ V(32x64)
      acc[t][0] = wmma_bf16(pa, vf[0], acc[t][0]);
      acc[t][1] = wmma_bf16(pa, vf[1], acc[t][1]);
      acc[t][2] = wmma_bf16(pa, vf[2], acc[t][2]);
      acc[t][3] = wmma_bf16(pa, vf[3], acc[t][3]);
    }
  }

  // epilogue: normalize and store ctx bf16 in [B, S, D] (heads re-interleaved)
  const int b = bh >> 3, h = bh & 7;
#pragma unroll
  for (int t = 0; t < 2; ++t) {
#pragma unroll
    for (int v = 0; v < 8; ++v) {
      const float inv = 1.0f / lrow[t][v];
      const size_t row =
          ((size_t)b * S_ + (size_t)qb * 32 + t * 16 + mh + v) * D_ + (size_t)h * DK_;
      Ctx[row + 0  + cn] = (__bf16)(acc[t][0][v] * inv);
      Ctx[row + 16 + cn] = (__bf16)(acc[t][1][v] * inv);
      Ctx[row + 32 + cn] = (__bf16)(acc[t][2][v] * inv);
      Ctx[row + 48 + cn] = (__bf16)(acc[t][3][v] * inv);
    }
  }
}

// ---------------------------------------------------------------------------
extern "C" void kernel_launch(void* const* d_in, const int* in_sizes, int n_in,
                              void* d_out, int out_size, void* d_ws, size_t ws_size,
                              hipStream_t stream) {
  const float* q  = (const float*)d_in[0];
  const float* k  = (const float*)d_in[1];
  const float* v  = (const float*)d_in[2];
  const float* Wq = (const float*)d_in[3];
  const float* bq = (const float*)d_in[4];
  const float* Wk = (const float*)d_in[5];
  const float* bk = (const float*)d_in[6];
  const float* Wv = (const float*)d_in[7];
  const float* bv = (const float*)d_in[8];
  const float* Wo = (const float*)d_in[9];
  const float* bo = (const float*)d_in[10];

  char* ws = (char*)d_ws;
  size_t off = 0;
  auto alloc = [&](size_t bytes) -> void* {
    void* p = ws + off;
    off += (bytes + 255) & ~(size_t)255;
    return p;
  };
  const size_t actE = (size_t)BS_ * D_;   // 4M elements per activation
  const size_t wgtE = (size_t)D_ * D_;    // 256K elements per weight

  __bf16* qb  = (__bf16*)alloc(actE * 2);
  __bf16* kb  = (__bf16*)alloc(actE * 2);
  __bf16* vb  = (__bf16*)alloc(actE * 2);
  __bf16* Wqb = (__bf16*)alloc(wgtE * 2);
  __bf16* Wkb = (__bf16*)alloc(wgtE * 2);
  __bf16* Wvb = (__bf16*)alloc(wgtE * 2);
  __bf16* Wob = (__bf16*)alloc(wgtE * 2);
  __bf16* Qhd = (__bf16*)alloc(actE * 2);  // [B,H,S,DK]
  __bf16* Khd = (__bf16*)alloc(actE * 2);  // [B,H,S,DK]
  __bf16* Vtd = (__bf16*)alloc(actE * 2);  // [B,H,DK,S]
  __bf16* Ctx = (__bf16*)alloc(actE * 2);  // [B,S,D]

  // Stage 0: conversions
  const int act4 = (int)(actE / 4), wgt4 = (int)(wgtE / 4);
  cvt_bf16<<<(act4 + 255) / 256, 256, 0, stream>>>(q,  qb,  act4);
  cvt_bf16<<<(act4 + 255) / 256, 256, 0, stream>>>(k,  kb,  act4);
  cvt_bf16<<<(act4 + 255) / 256, 256, 0, stream>>>(v,  vb,  act4);
  cvt_bf16<<<(wgt4 + 255) / 256, 256, 0, stream>>>(Wq, Wqb, wgt4);
  cvt_bf16<<<(wgt4 + 255) / 256, 256, 0, stream>>>(Wk, Wkb, wgt4);
  cvt_bf16<<<(wgt4 + 255) / 256, 256, 0, stream>>>(Wv, Wvb, wgt4);
  cvt_bf16<<<(wgt4 + 255) / 256, 256, 0, stream>>>(Wo, Wob, wgt4);

  // Stage 1: Q/K/V projections (one wave per 64x64 output tile)
  const int gemmBlocks = (BS_ / 64) * (D_ / 64) / 8;   // 128
  gemm512<<<gemmBlocks, 256, 0, stream>>>(qb, Wqb, bq, (void*)Qhd, 0);
  gemm512<<<gemmBlocks, 256, 0, stream>>>(kb, Wkb, bk, (void*)Khd, 0);
  gemm512<<<gemmBlocks, 256, 0, stream>>>(vb, Wvb, bv, (void*)Vtd, 1);

  // Stage 2: flash attention (B*H*(S/32) = 2048 waves / 8 per block)
  attn_fa<<<(B_ * H_ * (S_ / 32)) / 8, 256, 0, stream>>>(Qhd, Khd, Vtd, Ctx);

  // Stage 3: output projection, f32 to d_out
  gemm512<<<gemmBlocks, 256, 0, stream>>>(Ctx, Wob, bo, d_out, 2);
}
